// MoELayer_81183471829576
// MI455X (gfx1250) — matrix operand order repaired and checked by reading, not compiled, not run on hardware
//
#include <hip/hip_runtime.h>

#define NUM_EXPERTS 8
#define D_MODEL 1024
#define D_FF 4096
#define TOKENS 8192

typedef __attribute__((ext_vector_type(16))) _Float16 v16h;
typedef __attribute__((ext_vector_type(8)))  float    v8f;
typedef __attribute__((ext_vector_type(4)))  unsigned int u32x4;
typedef __attribute__((ext_vector_type(4)))  float    f32x4;
typedef __attribute__((ext_vector_type(4)))  int      i32x4;

// gcc-style vector to match the async-builtin's parameter type exactly
typedef int v4i_gcc __attribute__((vector_size(16)));
typedef __attribute__((address_space(1))) v4i_gcc* gptr_b128;
typedef __attribute__((address_space(3))) v4i_gcc* lptr_b128;

union Frag { v16h v; u32x4 q[2]; };

#define BM 128   // rows (tokens/slots) per block
#define BN 128   // output cols per block
#define BK 32    // K step (f16 WMMA K)
#define LDK 40   // BK + 8 pad (keeps 16B alignment, spreads banks)

#if __has_builtin(__builtin_amdgcn_global_load_async_to_lds_b128)
#define HAVE_ASYNC 1
#else
#define HAVE_ASYNC 0
#endif

static __device__ __forceinline__ void copy8_async(const _Float16* g, _Float16* l) {
#if HAVE_ASYNC
  __builtin_amdgcn_global_load_async_to_lds_b128(
      (gptr_b128)const_cast<_Float16*>(g), (lptr_b128)l, 0, 0);
#else
  *(u32x4*)l = *(const u32x4*)g;
#endif
}

static __device__ __forceinline__ void wait_async_done() {
#if HAVE_ASYNC
#if __has_builtin(__builtin_amdgcn_s_wait_asynccnt)
  __builtin_amdgcn_s_wait_asynccnt(0);
#else
  asm volatile("s_wait_asynccnt 0x0" ::: "memory");
#endif
#endif
}

static __device__ __forceinline__ v8f wmma_f16(Frag a, Frag b, v8f c) {
  return __builtin_amdgcn_wmma_f32_16x16x32_f16(false, a.v, false, b.v,
                                                (short)0, c, false, false);
}

// ---------------------------------------------------------------- zero counts
__global__ void moe_zero(int* __restrict__ counts) {
  if (threadIdx.x < NUM_EXPERTS) counts[threadIdx.x] = 0;
}

// ------------------------------------------------- gate + route + x -> f16
__global__ __launch_bounds__(256) void moe_gate(
    const float* __restrict__ x, const float* __restrict__ gw,
    const float* __restrict__ gb, _Float16* __restrict__ xb,
    int* __restrict__ etok, float* __restrict__ eprob,
    i32x4* __restrict__ route, int* __restrict__ counts) {
  int gid = blockIdx.x * blockDim.x + threadIdx.x;
  int t = gid >> 5;
  int lane = gid & 31;
  if (t >= TOKENS) return;
  const float* xr = x + (size_t)t * D_MODEL;
  float acc[NUM_EXPERTS];
#pragma unroll
  for (int e = 0; e < NUM_EXPERTS; ++e) acc[e] = 0.0f;
  for (int j = 0; j < D_MODEL / 32; ++j) {
    int d = j * 32 + lane;
    float xv = xr[d];
    xb[(size_t)t * D_MODEL + d] = (_Float16)xv;
#pragma unroll
    for (int e = 0; e < NUM_EXPERTS; ++e) acc[e] += xv * gw[d * NUM_EXPERTS + e];
  }
#pragma unroll
  for (int e = 0; e < NUM_EXPERTS; ++e) {
#pragma unroll
    for (int m = 16; m >= 1; m >>= 1) acc[e] += __shfl_xor(acc[e], m, 32);
  }
  if (lane == 0) {
    float mx = -1e30f;
#pragma unroll
    for (int e = 0; e < NUM_EXPERTS; ++e) { acc[e] += gb[e]; mx = fmaxf(mx, acc[e]); }
    float p[NUM_EXPERTS]; float s = 0.0f;
#pragma unroll
    for (int e = 0; e < NUM_EXPERTS; ++e) { p[e] = __expf(acc[e] - mx); s += p[e]; }
    float inv = 1.0f / s;
#pragma unroll
    for (int e = 0; e < NUM_EXPERTS; ++e) p[e] *= inv;
    int i0 = 0;
#pragma unroll
    for (int e = 1; e < NUM_EXPERTS; ++e) if (p[e] > p[i0]) i0 = e;
    int i1 = (i0 == 0) ? 1 : 0;
#pragma unroll
    for (int e = 0; e < NUM_EXPERTS; ++e) if (e != i0 && p[e] > p[i1]) i1 = e;
    int pos0 = atomicAdd(&counts[i0], 1);
    int pos1 = atomicAdd(&counts[i1], 1);
    etok[i0 * TOKENS + pos0] = t; eprob[i0 * TOKENS + pos0] = p[i0];
    etok[i1 * TOKENS + pos1] = t; eprob[i1 * TOKENS + pos1] = p[i1];
    i32x4 r; r.x = i0; r.y = pos0; r.z = i1; r.w = pos1;
    route[t] = r;
  }
}

// ------------------------------------------------------------- expert offsets
__global__ void moe_offsets(const int* __restrict__ counts, int* __restrict__ offs) {
  if (threadIdx.x == 0) {
    int s = 0;
    for (int e = 0; e < NUM_EXPERTS; ++e) { offs[e] = s; s += counts[e]; }
  }
}

// ---------------------- transpose + fp32->f16: src[R][C] f32 -> dst[C][R] f16
__global__ __launch_bounds__(256) void transpose_cvt(
    const float* __restrict__ src, _Float16* __restrict__ dst, int R, int C) {
  __shared__ float tile[32][33];
  const float* s = src + (size_t)blockIdx.z * R * C;
  _Float16* d = dst + (size_t)blockIdx.z * R * C;
  int r0 = blockIdx.y * 32, c0 = blockIdx.x * 32;
  int tx = threadIdx.x & 31, ty = threadIdx.x >> 5;  // 8 rows of 32 lanes
#pragma unroll
  for (int i = 0; i < 4; ++i)
    tile[ty + i * 8][tx] = s[(size_t)(r0 + ty + i * 8) * C + c0 + tx];
  __syncthreads();
#pragma unroll
  for (int i = 0; i < 4; ++i)
    d[(size_t)(c0 + ty + i * 8) * R + r0 + tx] = (_Float16)tile[tx][ty + i * 8];
}

// --------------------------------------- GEMM1: H = silu(gather(X) @ W1t + b1)
// A: gathered token rows of xb (f16, K-major). B: w1t rows (f16, [N][K]).
__global__ __launch_bounds__(256) void moe_gemm1(
    const _Float16* __restrict__ xb, const _Float16* __restrict__ w1t,
    const float* __restrict__ b1, const int* __restrict__ etok,
    const int* __restrict__ counts, const int* __restrict__ offs,
    _Float16* __restrict__ H) {
  const int e = blockIdx.z;
  const int Me = counts[e];
  const int mt = blockIdx.y;
  if (mt * BM >= Me) return;
  const int nt = blockIdx.x;
  const int n0 = nt * BN;

  __shared__ __align__(16) _Float16 As[2][BM * LDK];
  __shared__ __align__(16) _Float16 Bs[2][BN * LDK];

  const int tid = threadIdx.x;
  const int arow = tid >> 1;            // 0..127 (row of both 128x32 tiles)
  const int aseg = (tid & 1) * 16;      // k offset 0 or 16
  const int rr = mt * BM + arow;
  const int tokA = etok[e * TOKENS + (rr < Me ? rr : 0)];
  const _Float16* pA = xb + (size_t)tokA * D_MODEL;
  const _Float16* pB = w1t + (size_t)e * D_MODEL * D_FF + (size_t)(n0 + arow) * D_MODEL;

  const int lane = tid & 31;
  const int w = tid >> 5;
  const int wm = (w & 3) * 32;
  const int wn = (w >> 2) * 64;
  const int lh = lane >> 4;
  const int l16 = lane & 15;

  v8f zero = {0.f, 0.f, 0.f, 0.f, 0.f, 0.f, 0.f, 0.f};
  v8f acc[2][4] = {{zero, zero, zero, zero}, {zero, zero, zero, zero}};

  // prologue: stage 0
  copy8_async(pA + aseg,     &As[0][arow * LDK + aseg]);
  copy8_async(pA + aseg + 8, &As[0][arow * LDK + aseg + 8]);
  copy8_async(pB + aseg,     &Bs[0][arow * LDK + aseg]);
  copy8_async(pB + aseg + 8, &Bs[0][arow * LDK + aseg + 8]);

  const int NK = D_MODEL / BK;
  for (int ks = 0; ks < NK; ++ks) {
    wait_async_done();
    __syncthreads();
    if (ks + 1 < NK) {
      const int kk = (ks + 1) * BK;
      const int nb = (ks + 1) & 1;
      copy8_async(pA + kk + aseg,     &As[nb][arow * LDK + aseg]);
      copy8_async(pA + kk + aseg + 8, &As[nb][arow * LDK + aseg + 8]);
      copy8_async(pB + kk + aseg,     &Bs[nb][arow * LDK + aseg]);
      copy8_async(pB + kk + aseg + 8, &Bs[nb][arow * LDK + aseg + 8]);
    }
    const int cb = ks & 1;
    Frag a[2], b[4];
#pragma unroll
    for (int mi = 0; mi < 2; ++mi) {
      const _Float16* p = &As[cb][(wm + mi * 16 + l16) * LDK];
      a[mi].q[0] = *(const u32x4*)(p + lh * 8);
      a[mi].q[1] = *(const u32x4*)(p + 16 + lh * 8);
    }
#pragma unroll
    for (int ni = 0; ni < 4; ++ni) {
      const _Float16* p = &Bs[cb][(wn + ni * 16 + l16) * LDK];
      b[ni].q[0] = *(const u32x4*)(p + lh * 8);
      b[ni].q[1] = *(const u32x4*)(p + 16 + lh * 8);
    }
#pragma unroll
    for (int mi = 0; mi < 2; ++mi)
#pragma unroll
      for (int ni = 0; ni < 4; ++ni)
        acc[mi][ni] = wmma_f16(a[mi], b[ni], acc[mi][ni]);
  }

  const float* b1e = b1 + e * D_FF;
  const int base = offs[e] + mt * BM;
#pragma unroll
  for (int mi = 0; mi < 2; ++mi)
#pragma unroll
    for (int ni = 0; ni < 4; ++ni)
#pragma unroll
      for (int r = 0; r < 8; ++r) {
        int rowL = wm + mi * 16 + r + lh * 8;
        int colL = wn + ni * 16 + l16;
        if (mt * BM + rowL < Me) {
          float v = acc[mi][ni][r] + b1e[n0 + colL];
          v = v / (1.0f + __expf(-v));  // silu
          H[(size_t)(base + rowL) * D_FF + n0 + colL] = (_Float16)v;
        }
      }
}

// --------------------------------------- GEMM2: O = prob * (H @ W2t + b2)
__global__ __launch_bounds__(256) void moe_gemm2(
    const _Float16* __restrict__ H, const _Float16* __restrict__ w2t,
    const float* __restrict__ b2, const float* __restrict__ eprob,
    const int* __restrict__ counts, const int* __restrict__ offs,
    float* __restrict__ O) {
  const int e = blockIdx.z;
  const int Me = counts[e];
  const int mt = blockIdx.y;
  if (mt * BM >= Me) return;
  const int nt = blockIdx.x;
  const int n0 = nt * BN;

  __shared__ __align__(16) _Float16 As[2][BM * LDK];
  __shared__ __align__(16) _Float16 Bs[2][BN * LDK];

  const int tid = threadIdx.x;
  const int arow = tid >> 1;
  const int aseg = (tid & 1) * 16;
  const int rr = mt * BM + arow;
  const int rclamp = (rr < Me) ? rr : (Me - 1);
  const _Float16* pA = H + (size_t)(offs[e] + rclamp) * D_FF;
  const _Float16* pB = w2t + (size_t)e * D_FF * D_MODEL + (size_t)(n0 + arow) * D_FF;

  const int lane = tid & 31;
  const int w = tid >> 5;
  const int wm = (w & 3) * 32;
  const int wn = (w >> 2) * 64;
  const int lh = lane >> 4;
  const int l16 = lane & 15;

  v8f zero = {0.f, 0.f, 0.f, 0.f, 0.f, 0.f, 0.f, 0.f};
  v8f acc[2][4] = {{zero, zero, zero, zero}, {zero, zero, zero, zero}};

  copy8_async(pA + aseg,     &As[0][arow * LDK + aseg]);
  copy8_async(pA + aseg + 8, &As[0][arow * LDK + aseg + 8]);
  copy8_async(pB + aseg,     &Bs[0][arow * LDK + aseg]);
  copy8_async(pB + aseg + 8, &Bs[0][arow * LDK + aseg + 8]);

  const int NK = D_FF / BK;
  for (int ks = 0; ks < NK; ++ks) {
    wait_async_done();
    __syncthreads();
    if (ks + 1 < NK) {
      const int kk = (ks + 1) * BK;
      const int nb = (ks + 1) & 1;
      copy8_async(pA + kk + aseg,     &As[nb][arow * LDK + aseg]);
      copy8_async(pA + kk + aseg + 8, &As[nb][arow * LDK + aseg + 8]);
      copy8_async(pB + kk + aseg,     &Bs[nb][arow * LDK + aseg]);
      copy8_async(pB + kk + aseg + 8, &Bs[nb][arow * LDK + aseg + 8]);
    }
    const int cb = ks & 1;
    Frag a[2], b[4];
#pragma unroll
    for (int mi = 0; mi < 2; ++mi) {
      const _Float16* p = &As[cb][(wm + mi * 16 + l16) * LDK];
      a[mi].q[0] = *(const u32x4*)(p + lh * 8);
      a[mi].q[1] = *(const u32x4*)(p + 16 + lh * 8);
    }
#pragma unroll
    for (int ni = 0; ni < 4; ++ni) {
      const _Float16* p = &Bs[cb][(wn + ni * 16 + l16) * LDK];
      b[ni].q[0] = *(const u32x4*)(p + lh * 8);
      b[ni].q[1] = *(const u32x4*)(p + 16 + lh * 8);
    }
#pragma unroll
    for (int mi = 0; mi < 2; ++mi)
#pragma unroll
      for (int ni = 0; ni < 4; ++ni)
        acc[mi][ni] = wmma_f16(a[mi], b[ni], acc[mi][ni]);
  }

  const float* b2e = b2 + e * D_MODEL;
  const int base = offs[e] + mt * BM;
#pragma unroll
  for (int mi = 0; mi < 2; ++mi)
#pragma unroll
    for (int ni = 0; ni < 4; ++ni)
#pragma unroll
      for (int r = 0; r < 8; ++r) {
        int rowL = wm + mi * 16 + r + lh * 8;
        int colL = wn + ni * 16 + l16;
        if (mt * BM + rowL < Me) {
          float p = eprob[e * TOKENS + mt * BM + rowL];
          float v = (acc[mi][ni][r] + b2e[n0 + colL]) * p;
          O[(size_t)(base + rowL) * D_MODEL + n0 + colL] = v;
        }
      }
}

// --------------------------------------- combine the two slots of each token
__global__ __launch_bounds__(256) void moe_combine(
    const float* __restrict__ O, const i32x4* __restrict__ route,
    const int* __restrict__ offs, float* __restrict__ out) {
  unsigned idx = blockIdx.x * 256u + threadIdx.x;  // one float4 per thread
  int t = idx >> 8;
  int c = (idx & 255) * 4;
  i32x4 r = route[t];
  size_t s0 = (size_t)(offs[r.x] + r.y) * D_MODEL + c;
  size_t s1 = (size_t)(offs[r.z] + r.w) * D_MODEL + c;
  f32x4 a = *(const f32x4*)(O + s0);
  f32x4 b = *(const f32x4*)(O + s1);
  f32x4 csum = a + b;
  *(f32x4*)(out + (size_t)t * D_MODEL + c) = csum;
}

extern "C" void kernel_launch(void* const* d_in, const int* in_sizes, int n_in,
                              void* d_out, int out_size, void* d_ws, size_t ws_size,
                              hipStream_t stream) {
  (void)in_sizes; (void)n_in; (void)out_size; (void)ws_size;
  const float* x  = (const float*)d_in[0];
  const float* gw = (const float*)d_in[1];
  const float* gb = (const float*)d_in[2];
  const float* w1 = (const float*)d_in[3];
  const float* b1 = (const float*)d_in[4];
  const float* w2 = (const float*)d_in[5];
  const float* b2 = (const float*)d_in[6];
  float* out = (float*)d_out;

  char* ws = (char*)d_ws;
  size_t o = 0;
  _Float16* xb = (_Float16*)(ws + o); o += (size_t)TOKENS * D_MODEL * sizeof(_Float16);
  int*   etok  = (int*)(ws + o);      o += (size_t)NUM_EXPERTS * TOKENS * sizeof(int);
  float* eprob = (float*)(ws + o);    o += (size_t)NUM_EXPERTS * TOKENS * sizeof(float);
  i32x4* route = (i32x4*)(ws + o);    o += (size_t)TOKENS * sizeof(i32x4);
  int*   counts = (int*)(ws + o);     o += 256;
  int*   offs   = (int*)(ws + o);     o += 256;
  _Float16* H = (_Float16*)(ws + o);  o += (size_t)TOKENS * 2 * D_FF * sizeof(_Float16);
  float*    O = (float*)(ws + o);     o += (size_t)TOKENS * 2 * D_MODEL * sizeof(float);
  _Float16* w1t = (_Float16*)(ws + o); o += (size_t)NUM_EXPERTS * D_MODEL * D_FF * sizeof(_Float16);
  _Float16* w2t = (_Float16*)(ws + o); o += (size_t)NUM_EXPERTS * D_FF * D_MODEL * sizeof(_Float16);

  moe_zero<<<1, 64, 0, stream>>>(counts);
  moe_gate<<<TOKENS / 8, 256, 0, stream>>>(x, gw, gb, xb, etok, eprob, route, counts);
  moe_offsets<<<1, 32, 0, stream>>>(counts, offs);
  // w1 [1024][4096] -> w1t [4096][1024]; w2 [4096][1024] -> w2t [1024][4096]
  dim3 gt1(D_FF / 32, D_MODEL / 32, NUM_EXPERTS);
  transpose_cvt<<<gt1, 256, 0, stream>>>(w1, w1t, D_MODEL, D_FF);
  dim3 gt2(D_MODEL / 32, D_FF / 32, NUM_EXPERTS);
  transpose_cvt<<<gt2, 256, 0, stream>>>(w2, w2t, D_FF, D_MODEL);

  dim3 g1(D_FF / BN, TOKENS / BM, NUM_EXPERTS);
  moe_gemm1<<<g1, 256, 0, stream>>>(xb, w1t, b1, etok, counts, offs, H);
  dim3 g2(D_MODEL / BN, TOKENS / BM, NUM_EXPERTS);
  moe_gemm2<<<g2, 256, 0, stream>>>(H, w2t, b2, eprob, counts, offs, O);
  moe_combine<<<(TOKENS * D_MODEL / 4) / 256, 256, 0, stream>>>(O, route, offs, out);
}
